// DeformableAttention_37847251812558
// MI455X (gfx1250) — matrix operand order repaired and checked by reading, not compiled, not run on hardware
//
#include <hip/hip_runtime.h>
#include <hip/hip_bf16.h>

typedef __attribute__((ext_vector_type(16))) _Float16 v16h;
typedef __attribute__((ext_vector_type(8)))  _Float16 v8h;
typedef __attribute__((ext_vector_type(8)))  float    v8f;

#define HH   64
#define WW   64
#define DD   256
#define BB   2
#define NPIX 8192            // B*H*W
#define NHEAD 8
#define NPOINTS 4
#define COUT 96              // 64 offset channels + 32 attn channels

// --- gfx1250 async global->LDS copy helpers (ASYNCcnt-tracked) --------------
__device__ __forceinline__ void async_copy_b128(void* lds_dst, const void* gsrc) {
    unsigned           l = (unsigned)(size_t)lds_dst;
    unsigned long long g = (unsigned long long)(size_t)gsrc;
    asm volatile("global_load_async_to_lds_b128 %0, %1, off" :: "v"(l), "v"(g) : "memory");
}
__device__ __forceinline__ void wait_async0() {
#if __has_builtin(__builtin_amdgcn_s_wait_asynccnt)
    __builtin_amdgcn_s_wait_asynccnt(0);
#else
    asm volatile("s_wait_asynccnt 0" ::: "memory");
#endif
}

// ---------------------------------------------------------------------------
// Prep: fp32 -> f16 conversion (x and out_w; out_w (Dout,Din) row-major with
// Din contiguous == exactly the WMMA B-matrix per-lane-contiguous-K layout).
// ---------------------------------------------------------------------------
__global__ void da_cvt_f16(const float* __restrict__ in, _Float16* __restrict__ out, int n) {
    int i = blockIdx.x * blockDim.x + threadIdx.x;
    if (i < n) out[i] = (_Float16)in[i];
}

// ---------------------------------------------------------------------------
// Prep: pack conv weights (offset_w ++ attn_w, OIHW) into fragment-ready f16:
// wk[((hi*96 + m)*72 + kt)*16 + j]: the 16 halves lane (m%16 + 16*hi) of
// M-tile m/16 needs at K-step kt of v_wmma_f32_16x16x32_f16.
// K order within a row: k = tap*256 + cin, tap = ky*3+kx.
// ---------------------------------------------------------------------------
__global__ void da_pack_wconv(const float* __restrict__ offw, const float* __restrict__ attw,
                              _Float16* __restrict__ wk) {
    int i = blockIdx.x * blockDim.x + threadIdx.x;           // 2*96*72*16 = 221184
    if (i >= 2 * COUT * 72 * 16) return;
    int j  = i & 15;
    int kt = (i >> 4) % 72;
    int m  = (i >> 4) / 72 % COUT;
    int hi = i / (16 * 72 * COUT);
    int kin = j;                                             // A-frag K pattern (contiguous)
    int k  = kt * 32 + hi * 16 + kin;                        // global K 0..2303
    int tap = k >> 8, ci = k & 255;
    int ky = tap / 3, kx = tap % 3;
    float v = (m < 64) ? offw[(((size_t)m * 256 + ci) * 3 + ky) * 3 + kx]
                       : attw[(((size_t)(m - 64) * 256 + ci) * 3 + ky) * 3 + kx];
    wk[i] = (_Float16)v;
}

// ---------------------------------------------------------------------------
// Fused 3x3 conv (offset + attn heads) as im2col-free WMMA GEMM.
// Grid: 512 WGs (16-pixel N-tiles, all in one image row). Block: 192 = 6 waves
// = 6 M-tiles of the 96 output channels.  The WG's 3x18 input halo (27 KB f16)
// is async-staged into LDS once (zero-filled border), then all 72 WMMA B
// fragments come from ds_load_b128 — no redundant L2 traffic, no divergence
// in the inner loop.  convout layout: [pixel][96] fp32, bias added here.
// ---------------------------------------------------------------------------
__global__ __launch_bounds__(192) void da_conv_wmma(
        const _Float16* __restrict__ xh, const _Float16* __restrict__ wk,
        const float* __restrict__ offb, const float* __restrict__ attb,
        float* __restrict__ convout) {
    __shared__ _Float16 tile[3 * 18 * DD];   // 27648 B
    const int lane = threadIdx.x & 31;
    const int wv   = threadIdx.x >> 5;       // M-tile 0..5
    const int col  = lane & 15;              // A-row / B-col within tile
    const int hi   = lane >> 4;
    const int base = blockIdx.x * 16;        // first pixel of N-tile
    const int b    = base >> 12;
    const int py   = (base >> 6) & 63;
    const int px0  = base & 63;

    // ---- async stage: 54 pixel vectors x 512 B = 1728 x 16 B transfers -----
    for (int it = 0; it < 9; ++it) {
        int idx  = threadIdx.x + it * 192;   // 0..1727
        int slot = idx >> 5;                 // 0..53 : (row 0..2) * 18 + col 0..17
        int part = idx & 31;                 // 16B chunk within 512B vector
        int ry = py  + (slot / 18) - 1;
        int rx = px0 + (slot % 18) - 1;
        _Float16* ldst = &tile[slot * DD + part * 8];
        if (ry >= 0 && ry < HH && rx >= 0 && rx < WW) {
            async_copy_b128(ldst, xh + (size_t)((b << 12) | (ry << 6) | rx) * DD + part * 8);
        } else {
            v8h z = {};
            *(v8h*)ldst = z;                 // zero-pad border
        }
    }
    wait_async0();
    __syncthreads();

    // ---- 72 WMMAs: A from fragment-ready weights, B from LDS halo ----------
    const _Float16* ap = wk + ((size_t)(hi * COUT + wv * 16 + col) * 72) * 16;
    v8f acc = {};
    for (int tap = 0; tap < 9; ++tap) {
        const int dy = tap / 3, dx = tap % 3;                 // neighbor (dy-1, dx-1)
        const _Float16* bp = &tile[(dy * 18 + col + dx) * DD + hi * 16];
        const _Float16* at = ap + (size_t)tap * 8 * 16;
        __builtin_prefetch(at + 8 * 16, 0, 0);                // next tap's weights
#pragma unroll
        for (int ks = 0; ks < 8; ++ks) {
            v16h afrag = *(const v16h*)(at + ks * 16);        // 1x32B global (L2)
            v16h bfrag = *(const v16h*)(bp + ks * 32);        // 2x ds_load_b128
            acc = __builtin_amdgcn_wmma_f32_16x16x32_f16(
                    false, afrag, false, bfrag, (short)0, acc, false, false);
        }
    }
    // D layout: VGPR r -> channel wv*16 + hi*8 + r, pixel col; 8 contiguous f32.
    float bias[8];
#pragma unroll
    for (int r = 0; r < 8; ++r) {
        int ch = wv * 16 + hi * 8 + r;
        bias[r] = (ch < 64) ? offb[ch] : attb[ch - 64];
    }
    float* dst = convout + (size_t)(base + col) * COUT + wv * 16 + hi * 8;
    float4 s0 = { acc[0] + bias[0], acc[1] + bias[1], acc[2] + bias[2], acc[3] + bias[3] };
    float4 s1 = { acc[4] + bias[4], acc[5] + bias[5], acc[6] + bias[6], acc[7] + bias[7] };
    *(float4*)(dst)     = s0;
    *(float4*)(dst + 4) = s1;
}

// ---------------------------------------------------------------------------
// Deformable sampling + softmax + head-mean.  One WG per pixel, 256 threads =
// one channel each. Threads 0..7 build the 32 sample descriptors (attention
// softmax/8 folded with bilinear corner weights & validity) into LDS, then
// everyone does 32*4 coalesced channel gathers.  Writes yavg as f16 [pix][256].
// Reference index quirks reproduced: attn at (y,x), offsets at (x,y),
// gx = y/63 + 0.1*offx, gy = x/63 + 0.1*offy, ix = ((gx+1)*64-1)/2.
// ---------------------------------------------------------------------------
__global__ __launch_bounds__(256) void da_sample(
        const float* __restrict__ xin, const float* __restrict__ convout,
        _Float16* __restrict__ yh) {
    __shared__ float s_w[32][4];
    __shared__ int   s_p[32][4];
    const int pix = blockIdx.x;
    const int b = pix >> 12;
    const int y = (pix >> 6) & 63;
    const int x = pix & 63;
    const int t = threadIdx.x;

    if (t < NHEAD) {
        const int hh = t;
        const float* arow = convout + (size_t)pix * COUT + 64 + hh * 4;
        float a0 = arow[0], a1 = arow[1], a2 = arow[2], a3 = arow[3];
        float m = fmaxf(fmaxf(a0, a1), fmaxf(a2, a3));
        float e0 = __expf(a0 - m), e1 = __expf(a1 - m), e2 = __expf(a2 - m), e3 = __expf(a3 - m);
        float inv = 0.125f / (e0 + e1 + e2 + e3);           // fold 1/NHEAD mean
        float ew[4] = { e0 * inv, e1 * inv, e2 * inv, e3 * inv };
        const int pixT = (b << 12) | (x << 6) | y;          // transposed pixel for offsets
        const float* orow = convout + (size_t)pixT * COUT + hh * 8;
#pragma unroll
        for (int pp = 0; pp < 4; ++pp) {
            float offx = orow[pp * 2 + 0];
            float offy = orow[pp * 2 + 1];
            float gx = (float)y * (1.0f / 63.0f) + 0.1f * offx;
            float gy = (float)x * (1.0f / 63.0f) + 0.1f * offy;
            float ix = ((gx + 1.0f) * (float)WW - 1.0f) * 0.5f;
            float iy = ((gy + 1.0f) * (float)HH - 1.0f) * 0.5f;
            float x0f = floorf(ix), y0f = floorf(iy);
            int x0 = (int)x0f, y0 = (int)y0f;
            float fx = ix - x0f, fy = iy - y0f;
            int s = hh * 4 + pp;
#pragma unroll
            for (int c = 0; c < 4; ++c) {
                int cx = x0 + (c & 1), cy = y0 + (c >> 1);
                float cw = ((c & 1) ? fx : 1.0f - fx) * ((c >> 1) ? fy : 1.0f - fy);
                bool v = (cx >= 0) & (cx < WW) & (cy >= 0) & (cy < HH);
                int cxc = min(max(cx, 0), WW - 1), cyc = min(max(cy, 0), HH - 1);
                s_w[s][c] = v ? ew[pp] * cw : 0.0f;
                s_p[s][c] = (b << 12) | (cyc << 6) | cxc;
            }
        }
    }
    __syncthreads();

    float acc = 0.0f;
    const float* xb = xin + t;                               // channel t
#pragma unroll 4
    for (int s = 0; s < 32; ++s) {
#pragma unroll
        for (int c = 0; c < 4; ++c)
            acc = fmaf(s_w[s][c], xb[(size_t)s_p[s][c] * DD], acc);
    }
    yh[(size_t)pix * DD + t] = (_Float16)acc;
}

// ---------------------------------------------------------------------------
// Output projection: out[n][do] = sum_k yavg[n][k] * out_w[do][k] + out_b[do].
// Grid 512, block 256 = 8 waves.  The shared 16x256 A tile (8 KB) is
// async-staged into LDS once; each wave computes two 16x16 do-tiles
// (wv and wv+8), reusing each LDS A fragment for two WMMAs.
// ---------------------------------------------------------------------------
__global__ __launch_bounds__(256) void da_outgemm_wmma(
        const _Float16* __restrict__ yh, const _Float16* __restrict__ wouth,
        const float* __restrict__ outb, float* __restrict__ out) {
    __shared__ _Float16 atile[16 * DD];      // 8 KB
    const int lane = threadIdx.x & 31;
    const int wv   = threadIdx.x >> 5;
    const int col  = lane & 15;
    const int hi   = lane >> 4;
    const int ptile = blockIdx.x;            // 0..511

    // ---- async stage A tile: 512 x 16B transfers, 2 per thread -------------
    for (int it = 0; it < 2; ++it) {
        int idx  = threadIdx.x + it * 256;   // 0..511
        int row  = idx >> 5;                 // pixel row 0..15
        int part = idx & 31;                 // 16B chunk
        async_copy_b128(&atile[row * DD + part * 8],
                        yh + (size_t)(ptile * 16 + row) * DD + part * 8);
    }
    wait_async0();
    __syncthreads();

    const _Float16* arow  = &atile[col * DD];
    const _Float16* brow0 = wouth + (size_t)(wv * 16 + col) * DD + hi * 16;        // do-tile wv
    const _Float16* brow1 = wouth + (size_t)((wv + 8) * 16 + col) * DD + hi * 16;  // do-tile wv+8

    v8f acc0 = {}, acc1 = {};
#pragma unroll
    for (int ks = 0; ks < 8; ++ks) {
        v8h alo = *(const v8h*)(arow + ks * 32 + hi * 16);
        v8h ahi = *(const v8h*)(arow + ks * 32 + hi * 16 + 8);
        v16h afrag;
#pragma unroll
        for (int j = 0; j < 8; ++j) { afrag[j] = alo[j]; afrag[j + 8] = ahi[j]; }
        v16h b0 = *(const v16h*)(brow0 + ks * 32);
        v16h b1 = *(const v16h*)(brow1 + ks * 32);
        acc0 = __builtin_amdgcn_wmma_f32_16x16x32_f16(
                false, afrag, false, b0, (short)0, acc0, false, false);
        acc1 = __builtin_amdgcn_wmma_f32_16x16x32_f16(
                false, afrag, false, b1, (short)0, acc1, false, false);
    }
#pragma unroll
    for (int half = 0; half < 2; ++half) {
        const v8f& acc = half ? acc1 : acc0;
        const int dcol = (wv + half * 8) * 16 + col;
        const float bo = outb[dcol];
#pragma unroll
        for (int r = 0; r < 8; ++r) {
            int prow = ptile * 16 + hi * 8 + r;
            out[(size_t)prow * DD + dcol] = acc[r] + bo;
        }
    }
}

// ---------------------------------------------------------------------------
// Launch
// ---------------------------------------------------------------------------
extern "C" void kernel_launch(void* const* d_in, const int* in_sizes, int n_in,
                              void* d_out, int out_size, void* d_ws, size_t ws_size,
                              hipStream_t stream) {
    const float* x      = (const float*)d_in[0];   // (B, HW, D)
    const float* offw   = (const float*)d_in[1];   // (64, 256, 3, 3)
    const float* offb   = (const float*)d_in[2];   // (64)
    const float* attw   = (const float*)d_in[3];   // (32, 256, 3, 3)
    const float* attb   = (const float*)d_in[4];   // (32)
    const float* outw   = (const float*)d_in[5];   // (256, 256)
    const float* outb   = (const float*)d_in[6];   // (256)
    float* out = (float*)d_out;                    // (B, HW, D)

    char* ws = (char*)d_ws;
    size_t off = 0;
    _Float16* xh      = (_Float16*)(ws + off); off += (size_t)NPIX * DD * 2;          // 4 MB
    _Float16* wk      = (_Float16*)(ws + off); off += (size_t)2 * COUT * 72 * 16 * 2; // 432 KB
    _Float16* wouth   = (_Float16*)(ws + off); off += (size_t)DD * DD * 2;            // 128 KB
    float*    convout = (float*)   (ws + off); off += (size_t)NPIX * COUT * 4;        // 3 MB
    _Float16* yh      = (_Float16*)(ws + off); off += (size_t)NPIX * DD * 2;          // 4 MB
    (void)ws_size; (void)in_sizes; (void)n_in; (void)out_size;

    // Prep
    da_cvt_f16<<<(NPIX * DD + 255) / 256, 256, 0, stream>>>(x, xh, NPIX * DD);
    da_cvt_f16<<<(DD * DD + 255) / 256, 256, 0, stream>>>(outw, wouth, DD * DD);
    da_pack_wconv<<<(2 * COUT * 72 * 16 + 255) / 256, 256, 0, stream>>>(offw, attw, wk);

    // Fused offset+attention 3x3 conv (WMMA, LDS-staged halo)
    da_conv_wmma<<<NPIX / 16, 192, 0, stream>>>(xh, wk, offb, attb, convout);

    // Deformable bilinear sampling + softmax + head mean
    da_sample<<<NPIX, 256, 0, stream>>>(x, convout, yh);

    // Output projection (WMMA, LDS-staged A tile)
    da_outgemm_wmma<<<NPIX / 16, 256, 0, stream>>>(yh, wouth, outb, out);
}
